// ResidualSSM_85847806313232
// MI455X (gfx1250) — compile-verified, matrix operand order checked by reading, not run on hardware
//
#include <hip/hip_runtime.h>
#include <hip/hip_bf16.h>
#include <math.h>

// ---------------------------------------------------------------------------
// ResidualSSM for MI455X (gfx1250, wave32).
// All dense GEMMs use v_wmma_f32_16x16x32_f16 (f16 in, f32 accumulate):
// the big input GEMM is HBM-bound (~380MB total traffic @ 23.3TB/s), so the
// matrix path only has to keep pace with memory; f16 WMMA does that with
// minimal VGPR pressure. SSM conv/scan are tiny (T=12, N=8) latency kernels.
// ---------------------------------------------------------------------------

typedef __attribute__((ext_vector_type(16))) _Float16 v16h;
typedef __attribute__((ext_vector_type(8)))  float    v8f;

#define MTILE 128
#define KTILE 32
#define NCHUNK 64
#define LDSS 36   // padded row stride (halfs) to spread LDS banks

union FragU { v16h h; unsigned u[8]; };

// ISA 7.12.2: 16-bit A/B fragment K index for 32-bit reg v, lane-group grp.
__device__ __forceinline__ int frag_k(int v, int grp) {
    int k = (v < 4) ? (2 * v) : (16 + 2 * (v - 4));
    return k + 8 * grp;
}

__device__ __forceinline__ float wave_sum32(float v) {
#pragma unroll
    for (int m = 16; m >= 1; m >>= 1) v += __shfl_xor(v, m, 32);
    return v;
}

// C[M x N] = concat_k(A0[.,K0], A1[.,K1]) @ W[N x K]^T + bias
// Block: 256 threads (8 waves). Tile: 128 rows x 64 cols. grid.y over N chunks.
__global__ __launch_bounds__(256)
void wmma_gemm_kernel(const float* __restrict__ A0, int lda0, int K0,
                      const float* __restrict__ A1, int lda1, int K1,
                      const float* __restrict__ W, const float* __restrict__ bias,
                      float* __restrict__ C, int N)
{
    __shared__ __align__(16) _Float16 lds_a[MTILE * LDSS];
    __shared__ __align__(16) _Float16 lds_w[NCHUNK * LDSS];

    const int tid  = threadIdx.x;
    const int wave = tid >> 5;
    const int lane = tid & 31;
    const int grp  = lane >> 4;
    const int lcol = lane & 15;
    const int rowBase = blockIdx.x * MTILE;
    const int ncol0   = blockIdx.y * NCHUNK;
    const int K = K0 + K1;

    v8f acc[4];
#pragma unroll
    for (int i = 0; i < 4; ++i)
#pragma unroll
        for (int j = 0; j < 8; ++j) acc[i][j] = 0.0f;

    // A loader mapping: each thread loads one 16-wide half-row per K step
    const int arow = tid >> 1;
    const int akh  = (tid & 1) * 16;
    const long grow = rowBase + arow;
    // W loader mapping: 64 n-rows x 32 k, 8 values per thread
    const int wn = tid >> 2;
    const int wkq = (tid & 3) * 8;

    for (int kb = 0; kb < K; kb += KTILE) {
        // ---- stage A tile (fp32 -> f16) ----
        {
            _Float16* dst = &lds_a[arow * LDSS + akh];
            const int kbase = kb + akh;
            if (kbase + 16 <= K0) {
                const float4* p = (const float4*)(A0 + grow * lda0 + kbase);
#pragma unroll
                for (int q = 0; q < 4; ++q) {
                    float4 f = p[q];
                    dst[q * 4 + 0] = (_Float16)f.x; dst[q * 4 + 1] = (_Float16)f.y;
                    dst[q * 4 + 2] = (_Float16)f.z; dst[q * 4 + 3] = (_Float16)f.w;
                }
            } else if (kbase >= K0 && kbase + 16 <= K) {
                const float2* p = (const float2*)(A1 + grow * lda1 + (kbase - K0));
#pragma unroll
                for (int q = 0; q < 8; ++q) {
                    float2 f = p[q];
                    dst[q * 2 + 0] = (_Float16)f.x; dst[q * 2 + 1] = (_Float16)f.y;
                }
            } else {
#pragma unroll
                for (int q = 0; q < 16; ++q) {
                    int k = kbase + q; float v = 0.0f;
                    if (k < K0) v = A0[grow * lda0 + k];
                    else if (k < K) v = A1[grow * lda1 + (k - K0)];
                    dst[q] = (_Float16)v;
                }
            }
        }
        // ---- stage W tile (fp32 -> f16) ----
        {
            const int n = ncol0 + wn;
            _Float16* dst = &lds_w[wn * LDSS + wkq];
            const int kbase = kb + wkq;
            if (n < N && kbase + 8 <= K) {
                const float2* p = (const float2*)(W + (long)n * K + kbase);
#pragma unroll
                for (int q = 0; q < 4; ++q) {
                    float2 f = p[q];
                    dst[q * 2 + 0] = (_Float16)f.x; dst[q * 2 + 1] = (_Float16)f.y;
                }
            } else {
#pragma unroll
                for (int q = 0; q < 8; ++q) {
                    int k = kbase + q;
                    float v = (n < N && k < K) ? W[(long)n * K + k] : 0.0f;
                    dst[q] = (_Float16)v;
                }
            }
        }
        __syncthreads();

        FragU af;
#pragma unroll
        for (int v = 0; v < 8; ++v)
            af.u[v] = *(const unsigned*)&lds_a[(wave * 16 + lcol) * LDSS + frag_k(v, grp)];
#pragma unroll
        for (int nt = 0; nt < 4; ++nt) {
            FragU bf;
#pragma unroll
            for (int v = 0; v < 8; ++v)
                bf.u[v] = *(const unsigned*)&lds_w[(nt * 16 + lcol) * LDSS + frag_k(v, grp)];
            acc[nt] = __builtin_amdgcn_wmma_f32_16x16x32_f16(
                false, af.h, false, bf.h, (short)0, acc[nt], false, false);
        }
        __syncthreads();
    }

    // ---- store (C/D layout: lane row split, 8 M-rows per lane) ----
    const int r0 = rowBase + wave * 16 + 8 * grp;
#pragma unroll
    for (int nt = 0; nt < 4; ++nt) {
        const int n = ncol0 + nt * 16 + lcol;
        if (n < N) {
            const float bv = bias ? bias[n] : 0.0f;
#pragma unroll
            for (int r = 0; r < 8; ++r)
                C[(long)(r0 + r) * N + n] = acc[nt][r] + bv;
        }
    }
}

// meta[b, d] = [site_tab[sid]; hour_tab[hid]](16) . W_meta[d]  + b_meta[d]
__global__ void meta_kernel(const float* __restrict__ site_tab,
                            const float* __restrict__ hour_tab,
                            const float* __restrict__ Wm, const float* __restrict__ bm,
                            const int* __restrict__ site_ids, const int* __restrict__ hours,
                            float* __restrict__ meta, int B)
{
    int idx = blockIdx.x * blockDim.x + threadIdx.x;
    int b = idx >> 6, d = idx & 63;
    if (b >= B) return;
    int sid = min(max(site_ids[b], 0), 19);
    int hid = min(max(hours[b], 0), 23);
    float acc = bm[d];
#pragma unroll
    for (int k = 0; k < 8; ++k) acc += site_tab[sid * 8 + k] * Wm[d * 16 + k];
#pragma unroll
    for (int k = 0; k < 8; ++k) acc += hour_tab[hid * 8 + k] * Wm[d * 16 + 8 + k];
    meta[(long)b * 64 + d] = acc;
}

// h = gelu(LN(x)) + pos_enc[t] + meta[b]   (D = 64, one wave per row)
__global__ __launch_bounds__(256)
void ln_gelu_kernel(const float* __restrict__ X, const float* __restrict__ g,
                    const float* __restrict__ bb, const float* __restrict__ pos,
                    const float* __restrict__ meta, float* __restrict__ out, int T)
{
    const int wave = threadIdx.x >> 5, lane = threadIdx.x & 31;
    const long row = (long)blockIdx.x * 8 + wave;
    const int t = (int)(row % T); const long bidx = row / T;
    float x0 = X[row * 64 + lane], x1 = X[row * 64 + 32 + lane];
    float m = wave_sum32(x0 + x1) * (1.0f / 64.0f);
    float d0 = x0 - m, d1 = x1 - m;
    float var = wave_sum32(d0 * d0 + d1 * d1) * (1.0f / 64.0f);
    float inv = rsqrtf(var + 1e-5f);
#pragma unroll
    for (int e = 0; e < 2; ++e) {
        int dd = lane + 32 * e;
        float y = (e ? d1 : d0) * inv * g[dd] + bb[dd];
        float gel = 0.5f * y * (1.0f + erff(y * 0.70710678118654752f));
        out[row * 64 + dd] = gel + pos[t * 64 + dd] + meta[bidx * 64 + dd];
    }
}

// out = LN(x + res)
__global__ __launch_bounds__(256)
void ln_res_kernel(const float* __restrict__ X, const float* __restrict__ R,
                   const float* __restrict__ g, const float* __restrict__ bb,
                   float* __restrict__ out)
{
    const int wave = threadIdx.x >> 5, lane = threadIdx.x & 31;
    const long row = (long)blockIdx.x * 8 + wave;
    float x0 = X[row * 64 + lane] + R[row * 64 + lane];
    float x1 = X[row * 64 + 32 + lane] + R[row * 64 + 32 + lane];
    float m = wave_sum32(x0 + x1) * (1.0f / 64.0f);
    float d0 = x0 - m, d1 = x1 - m;
    float var = wave_sum32(d0 * d0 + d1 * d1) * (1.0f / 64.0f);
    float inv = rsqrtf(var + 1e-5f);
    out[row * 64 + lane]      = d0 * inv * g[lane] + bb[lane];
    out[row * 64 + 32 + lane] = d1 * inv * g[lane + 32] + bb[lane + 32];
}

// depthwise causal conv (dconv=4) + SiLU; `reverse` implements flipped time.
__global__ void conv_silu_kernel(const float* __restrict__ Xs, const float* __restrict__ convw,
                                 const float* __restrict__ convb, float* __restrict__ XC,
                                 int T, int reverse)
{
    const int b = blockIdx.x, d = threadIdx.x;
    float w[4];
#pragma unroll
    for (int j = 0; j < 4; ++j) w[j] = convw[d * 4 + j];
    const float cb = convb[d];
    for (int tl = 0; tl < T; ++tl) {
        float acc = cb;
#pragma unroll
        for (int j = 0; j < 4; ++j) {
            int src = tl - 3 + j;
            if (src >= 0) {
                int sp = reverse ? (T - 1 - src) : src;
                acc += w[j] * Xs[((long)b * T + sp) * 64 + d];
            }
        }
        acc = acc / (1.0f + expf(-acc));   // SiLU
        int tp = reverse ? (T - 1 - tl) : tl;
        XC[((long)b * T + tp) * 64 + d] = acc;
    }
}

// pack [Wdt;WB;WC] -> Wcat(80x64), [bdt;0] -> bcat(80) for one fused GEMM
__global__ void pack_dtbc_kernel(const float* __restrict__ Wdt, const float* __restrict__ WB,
                                 const float* __restrict__ WC, const float* __restrict__ bdt,
                                 float* __restrict__ Wcat, float* __restrict__ bcat)
{
    int idx = blockIdx.x * blockDim.x + threadIdx.x;
    if (idx < 80 * 64) {
        int n = idx >> 6, k = idx & 63;
        float v;
        if (n < 64) v = Wdt[n * 64 + k];
        else if (n < 72) v = WB[(n - 64) * 64 + k];
        else v = WC[(n - 72) * 64 + k];
        Wcat[idx] = v;
    }
    if (idx < 80) bcat[idx] = (idx < 64) ? bdt[idx] : 0.0f;
}

// recurrent scan over T (state h[D][N] per batch), thread = channel d.
__global__ void ssm_scan_kernel(const float* __restrict__ DTBC, const float* __restrict__ XC,
                                const float* __restrict__ Xin, const float* __restrict__ Alog,
                                const float* __restrict__ Dp, float* __restrict__ Ycat,
                                int T, int reverse, int outOffset)
{
    const int b = blockIdx.x, d = threadIdx.x;
    __shared__ float sB[8], sC[8];
    float A[8], hst[8];
#pragma unroll
    for (int n = 0; n < 8; ++n) { A[n] = -expf(Alog[d * 8 + n]); hst[n] = 0.0f; }
    const float dp = Dp[d];
    for (int s = 0; s < T; ++s) {
        const int t = reverse ? (T - 1 - s) : s;
        const long row = (long)b * T + t;
        if (d < 8)           sB[d]     = DTBC[row * 80 + 64 + d];
        else if (d < 16)     sC[d - 8] = DTBC[row * 80 + 72 + (d - 8)];
        __syncthreads();
        float dpre = DTBC[row * 80 + d];
        float dt = (dpre > 20.0f) ? dpre : log1pf(expf(dpre));   // softplus
        float x = XC[row * 64 + d];
        float y = 0.0f;
#pragma unroll
        for (int n = 0; n < 8; ++n) {
            float dA = expf(A[n] * dt);
            hst[n] = hst[n] * dA + x * dt * sB[n];
            y += hst[n] * sC[n];
        }
        Ycat[row * 128 + outOffset + d] = y + Xin[row * 64 + d] * dp;
        __syncthreads();
    }
}

extern "C" void kernel_launch(void* const* d_in, const int* in_sizes, int n_in,
                              void* d_out, int out_size, void* d_ws, size_t ws_size,
                              hipStream_t stream)
{
    (void)in_sizes; (void)n_in; (void)out_size; (void)ws_size;
    const int B = 4096, T = 12, Din = 1536, Ds = 234, Cout = 234;
    const long Mr = (long)B * T;   // 49152

    const float* emb        = (const float*)d_in[0];
    const float* first_pass = (const float*)d_in[1];
    const float* W_in   = (const float*)d_in[2];
    const float* b_in   = (const float*)d_in[3];
    const float* ln_in_g = (const float*)d_in[4];
    const float* ln_in_b = (const float*)d_in[5];
    const float* site_tab = (const float*)d_in[6];
    const float* hour_tab = (const float*)d_in[7];
    const float* W_meta = (const float*)d_in[8];
    const float* b_meta = (const float*)d_in[9];
    const float* pos_enc = (const float*)d_in[10];
    const float* W_merge = (const float*)d_in[11];
    const float* b_merge = (const float*)d_in[12];
    const float* ln_g = (const float*)d_in[13];
    const float* ln_b = (const float*)d_in[14];
    const float* W_out = (const float*)d_in[15];
    const float* b_out = (const float*)d_in[16];
    const int* site_ids = (const int*)d_in[17];
    const int* hours    = (const int*)d_in[18];

    float* ws = (float*)d_ws;
    size_t off = 0;
    float* h0   = ws + off; off += (size_t)Mr * 64;
    float* h    = ws + off; off += (size_t)Mr * 64;
    float* meta = ws + off; off += (size_t)B * 64;
    float* Xs   = ws + off; off += (size_t)Mr * 64;
    float* XC   = ws + off; off += (size_t)Mr * 64;
    float* DTBC = ws + off; off += (size_t)Mr * 80;
    float* Ycat = ws + off; off += (size_t)Mr * 128;
    float* hm   = ws + off; off += (size_t)Mr * 64;
    float* hln  = ws + off; off += (size_t)Mr * 64;
    float* Wcat = ws + off; off += 80 * 64;
    float* bcat = ws + off; off += 80;

    const dim3 blk(256);
    const int mgrid = (int)(Mr / MTILE);   // 384

    // 1) h0 = concat(emb, first_pass) @ W_in^T + b_in   (K = 1770)
    wmma_gemm_kernel<<<dim3(mgrid, 1), blk, 0, stream>>>(
        emb, Din, Din, first_pass, Ds, Ds, W_in, b_in, h0, 64);
    // 2) per-batch meta projection
    meta_kernel<<<dim3((B * 64) / 256), blk, 0, stream>>>(
        site_tab, hour_tab, W_meta, b_meta, site_ids, hours, meta, B);
    // 3) h = gelu(LN(h0)) + pos + meta   (also the residual)
    ln_gelu_kernel<<<dim3((int)(Mr / 8)), blk, 0, stream>>>(
        h0, ln_in_g, ln_in_b, pos_enc, meta, h, T);

    // 4) bidirectional selective SSM (z-gate half of Wxz is unused by the ref)
    for (int dir = 0; dir < 2; ++dir) {
        const int base = 19 + dir * 9;
        const float* Wxz   = (const float*)d_in[base + 0];
        const float* convw = (const float*)d_in[base + 1];
        const float* convb = (const float*)d_in[base + 2];
        const float* Wdt   = (const float*)d_in[base + 3];
        const float* bdt   = (const float*)d_in[base + 4];
        const float* Alog  = (const float*)d_in[base + 5];
        const float* Dp    = (const float*)d_in[base + 6];
        const float* WB    = (const float*)d_in[base + 7];
        const float* WC    = (const float*)d_in[base + 8];

        wmma_gemm_kernel<<<dim3(mgrid, 1), blk, 0, stream>>>(
            h, 64, 64, nullptr, 0, 0, Wxz, nullptr, Xs, 64);          // x_ssm
        conv_silu_kernel<<<dim3(B), dim3(64), 0, stream>>>(Xs, convw, convb, XC, T, dir);
        pack_dtbc_kernel<<<dim3(20), blk, 0, stream>>>(Wdt, WB, WC, bdt, Wcat, bcat);
        wmma_gemm_kernel<<<dim3(mgrid, 2), blk, 0, stream>>>(
            XC, 64, 64, nullptr, 0, 0, Wcat, bcat, DTBC, 80);         // [dt|B|C]
        ssm_scan_kernel<<<dim3(B), dim3(64), 0, stream>>>(
            DTBC, XC, h, Alog, Dp, Ycat, T, dir, dir * 64);
    }

    // 5) merge: concat(h_f, h_b) @ W_merge^T + b_merge  (K = 128)
    wmma_gemm_kernel<<<dim3(mgrid, 1), blk, 0, stream>>>(
        Ycat, 128, 128, nullptr, 0, 0, W_merge, b_merge, hm, 64);
    // 6) LN(hm + res)
    ln_res_kernel<<<dim3((int)(Mr / 8)), blk, 0, stream>>>(hm, h, ln_g, ln_b, hln);
    // 7) out = hln @ W_out^T + b_out  (N = 234, grid.y covers 4x64 cols)
    wmma_gemm_kernel<<<dim3(mgrid, 4), blk, 0, stream>>>(
        hln, 64, 64, nullptr, 0, 0, W_out, b_out, (float*)d_out, Cout);
}
